// InteractionBlock_57707180589099
// MI455X (gfx1250) — matrix operand order repaired and checked by reading, not compiled, not run on hardware
//
#include <hip/hip_runtime.h>
#include <hip/hip_bf16.h>

typedef __attribute__((ext_vector_type(16))) _Float16 v16h;
typedef __attribute__((ext_vector_type(8)))  float    v8f;

#define E_EDGES 200000
#define T_TRIP  500000
#define LDSP    136   // LDS row stride in halfs for 16x128 tile staging

// silu without IEEE divide: v * rcp(1+exp(-v)); v_rcp_f32 is ~1ulp, fine here.
__device__ __forceinline__ float silu_f(float v) {
    return v * __builtin_amdgcn_rcpf(1.0f + __expf(-v));
}

__device__ __forceinline__ v8f zero8() {
    v8f z;
#pragma unroll
    for (int i = 0; i < 8; ++i) z[i] = 0.0f;
    return z;
}

__device__ __forceinline__ unsigned pack2h(float a, float b) {
    union { _Float16 h[2]; unsigned u; } p;
    p.h[0] = (_Float16)a; p.h[1] = (_Float16)b;
    return p.u;
}

// D = A(16x32 f16) * B(32x16 f16) + C, f32 accumulate
__device__ __forceinline__ v8f wmma16(v16h a, v16h b, v8f c) {
    return __builtin_amdgcn_wmma_f32_16x16x32_f16(false, a, false, b,
                                                  (short)0, c, false, false);
}

// A-fragment (16x32, f16) from row-major f32 source. lane: m=lane&15, q=lane>>4.
// VGPR r holds K = (r>=4)*16 + (r&3)*2 + q*8 + {0,1}
__device__ __forceinline__ v16h load_a_f32(const float* base, int ld, int row,
                                           int kbase, int q) {
    union { v16h v; unsigned u[8]; } f;
    const float* p = base + (size_t)row * ld + kbase + q * 8;
#pragma unroll
    for (int r = 0; r < 8; ++r) {
        int k = ((r & 4) << 2) + ((r & 3) << 1);
        float2 t = *(const float2*)(p + k);
        f.u[r] = pack2h(t.x, t.y);
    }
    return f.v;
}

// A-fragment from row-major f16 source (global or LDS-generic)
__device__ __forceinline__ v16h load_a_f16(const _Float16* base, int ld, int row,
                                           int kbase, int q) {
    union { v16h v; unsigned u[8]; } f;
    const _Float16* p = base + (size_t)row * ld + kbase + q * 8;
#pragma unroll
    for (int r = 0; r < 8; ++r) {
        int k = ((r & 4) << 2) + ((r & 3) << 1);
        f.u[r] = *(const unsigned*)(p + k);
    }
    return f.v;
}

// B-fragment (32x16, f16) from row-major [K x N] f16 weights: column n = lane&15,
// element s holds K = (s>=8)*16 + (lane>>4)*8 + (s&7).
__device__ __forceinline__ v16h load_b(const _Float16* w, int ld, int kbase,
                                       int n0, int lane) {
    v16h b;
    const int n = lane & 15, h = lane >> 4;
#pragma unroll
    for (int s = 0; s < 16; ++s) {
        int k = ((s >> 3) << 4) + h * 8 + (s & 7);
        b[s] = w[(size_t)(kbase + k) * ld + n0 + n];
    }
    return b;
}

// ---------------------------------------------------------------------------
// Kernel 1: weight conversion / repacking to f16
// ---------------------------------------------------------------------------
__global__ void prep_weights(const float* __restrict__ lin_ji_w,
                             const float* __restrict__ lin_kj_w,
                             const float* __restrict__ Wbig,
                             const float* __restrict__ before_w,
                             const float* __restrict__ lin_w,
                             const float* __restrict__ after_w,
                             _Float16* __restrict__ wji16,
                             _Float16* __restrict__ wkj16,
                             _Float16* __restrict__ Wr16,
                             _Float16* __restrict__ w3) {
    int i = blockIdx.x * blockDim.x + threadIdx.x;
    if (i < 16384) {
        wji16[i] = (_Float16)lin_ji_w[i];
        wkj16[i] = (_Float16)lin_kj_w[i];
    }
    if (i < 7 * 16384) {
        int mtx = i / 16384, o = i - mtx * 16384;
        float v;
        if (mtx == 0)      v = before_w[o];
        else if (mtx == 1) v = before_w[16384 + o];
        else if (mtx == 2) v = lin_w[o];
        else               v = after_w[(mtx - 3) * 16384 + o];
        w3[i] = (_Float16)v;
    }
    if (i < 131072) {
        // Wr[l, j*128+i] = W[i, j, l]
        int l = i >> 10, rest = i & 1023, j = rest >> 7, ii = rest & 127;
        Wr16[i] = (_Float16)Wbig[(ii * 8 + j) * 128 + l];
    }
}

// ---------------------------------------------------------------------------
// Kernel 2: rbf_p = rbf @ lin_rbf_w ; sbf_p = sbf @ lin_sbf_w ; agg = 0
// ---------------------------------------------------------------------------
__global__ void proj_small(const float* __restrict__ rbf,
                           const float* __restrict__ lin_rbf_w,
                           const float* __restrict__ sbf,
                           const float* __restrict__ lin_sbf_w,
                           float* __restrict__ rbf_p,
                           float* __restrict__ sbf_p,
                           float* __restrict__ agg) {
    int i = blockIdx.x * blockDim.x + threadIdx.x;
    if (i < E_EDGES * 128) {
        int e = i >> 7, n = i & 127;
        float s = 0.0f;
#pragma unroll
        for (int c = 0; c < 6; ++c) s += rbf[e * 6 + c] * lin_rbf_w[c * 128 + n];
        rbf_p[i] = s;
        agg[i] = 0.0f;
    }
    if (i < T_TRIP * 8) {
        int t = i >> 3, j = i & 7;
        float s = 0.0f;
#pragma unroll
        for (int c = 0; c < 42; ++c) s += sbf[t * 42 + c] * lin_sbf_w[c * 8 + j];
        sbf_p[i] = s;
    }
}

// ---------------------------------------------------------------------------
// Kernel 3: x_ji = silu(x@Wji + b), x_kj = silu(x@Wkj + b) * rbf_p  (f16 out)
// One wave per 16-row tile, WG = 128 threads = 4 waves.
// ---------------------------------------------------------------------------
__global__ __launch_bounds__(128) void edge_pre(
    const float* __restrict__ x, const float* __restrict__ rbf_p,
    const _Float16* __restrict__ wji, const _Float16* __restrict__ wkj,
    const float* __restrict__ bji, const float* __restrict__ bkj,
    float* __restrict__ x_ji, _Float16* __restrict__ x_kj) {
    const int lane = threadIdx.x & 31, wave = threadIdx.x >> 5;
    const int tile = blockIdx.x * 4 + wave;
    if (tile >= E_EDGES / 16) return;
    const int e0 = tile * 16;
    const int m = lane & 15, q = lane >> 4, col = m;
    const int row = e0 + m;

    v16h a[4];
#pragma unroll
    for (int ks = 0; ks < 4; ++ks) a[ks] = load_a_f32(x, 128, row, ks * 32, q);

#pragma unroll
    for (int nb = 0; nb < 8; ++nb) {
        const int n0 = nb * 16;
        // batch all 8 B-fragment loads, then run the WMMA chain
        v16h bj[4], bk[4];
#pragma unroll
        for (int ks = 0; ks < 4; ++ks) {
            bj[ks] = load_b(wji, 128, ks * 32, n0, lane);
            bk[ks] = load_b(wkj, 128, ks * 32, n0, lane);
        }
        v8f cji = zero8(), ckj = zero8();
#pragma unroll
        for (int ks = 0; ks < 4; ++ks) {
            cji = wmma16(a[ks], bj[ks], cji);
            ckj = wmma16(a[ks], bk[ks], ckj);
        }
        const float biasj = bji[n0 + col];
        const float biask = bkj[n0 + col];
#pragma unroll
        for (int r = 0; r < 8; ++r) {
            int rr = e0 + q * 8 + r;
            size_t o = (size_t)rr * 128 + n0 + col;
            x_ji[o] = silu_f(cji[r] + biasj);
            x_kj[o] = (_Float16)(silu_f(ckj[r] + biask) * rbf_p[o]);
        }
    }
}

// ---------------------------------------------------------------------------
// Kernel 4: the big one. Per 16-triplet tile:
//   g = x_kj[idx_kj]  (gathered f16 rows, L2-resident)
//   t[:, j*128+i] = g @ Wr   (f16 WMMA, f32 acc)
//   m[:, i] += sbf_p[:, j] * t
//   agg[idx_ji] += m  (global_atomic_add_f32)
// B-fragments are software-pipelined one block deep.
// ---------------------------------------------------------------------------
__global__ __launch_bounds__(128) void triplet_gemm(
    const _Float16* __restrict__ x_kj, const _Float16* __restrict__ Wr,
    const float* __restrict__ sbf_p, const int* __restrict__ idx_kj,
    const int* __restrict__ idx_ji, float* __restrict__ agg) {
    const int lane = threadIdx.x & 31, wave = threadIdx.x >> 5;
    const int tile = blockIdx.x * 4 + wave;
    if (tile >= T_TRIP / 16) return;
    const int t0 = tile * 16;
    const int m = lane & 15, q = lane >> 4, col = m;

    const int rowA = idx_kj[t0 + m];
    v16h a[4];
#pragma unroll
    for (int ks = 0; ks < 4; ++ks) a[ks] = load_a_f16(x_kj, 128, rowA, ks * 32, q);

    v8f macc[8];
#pragma unroll
    for (int ib = 0; ib < 8; ++ib) macc[ib] = zero8();

    // prologue: first B block (j=0, ib=0)
    v16h bf[4];
#pragma unroll
    for (int ks = 0; ks < 4; ++ks) bf[ks] = load_b(Wr, 1024, ks * 32, 0, lane);

    for (int j = 0; j < 8; ++j) {
        float sc[8];
#pragma unroll
        for (int r = 0; r < 8; ++r) sc[r] = sbf_p[(size_t)(t0 + q * 8 + r) * 8 + j];
#pragma unroll
        for (int ib = 0; ib < 8; ++ib) {
            v16h cur[4];
#pragma unroll
            for (int ks = 0; ks < 4; ++ks) cur[ks] = bf[ks];
            // prefetch next B block while current WMMAs run
            if (!(j == 7 && ib == 7)) {
                const int jn  = (ib == 7) ? j + 1 : j;
                const int ibn = (ib == 7) ? 0 : ib + 1;
                const int n0n = jn * 128 + ibn * 16;
#pragma unroll
                for (int ks = 0; ks < 4; ++ks)
                    bf[ks] = load_b(Wr, 1024, ks * 32, n0n, lane);
            }
            v8f t = zero8();
#pragma unroll
            for (int ks = 0; ks < 4; ++ks) t = wmma16(a[ks], cur[ks], t);
#pragma unroll
            for (int r = 0; r < 8; ++r) macc[ib][r] += sc[r] * t[r];
        }
    }

#pragma unroll
    for (int r = 0; r < 8; ++r) {
        const int rowO = idx_ji[t0 + q * 8 + r];
#pragma unroll
        for (int ib = 0; ib < 8; ++ib) {
            float* addr = agg + (size_t)rowO * 128 + ib * 16 + col;
            float val = macc[ib][r];
            asm volatile("global_atomic_add_f32 %0, %1, off"
                         :: "v"(addr), "v"(val) : "memory");
        }
    }
}

// ---------------------------------------------------------------------------
// Kernel 5 helpers: one 128x128 GEMM step with LDS f16 staging of the tile.
// ---------------------------------------------------------------------------
__device__ __forceinline__ void gemm128(const v8f hin[8], v8f hout[8],
                                        _Float16* lds, const _Float16* w,
                                        const float* bias, int lane) {
    const int m = lane & 15, q = lane >> 4, col = m;
    // stage C-layout tile into LDS as row-major f16 [16][LDSP]
#pragma unroll
    for (int ib = 0; ib < 8; ++ib)
#pragma unroll
        for (int r = 0; r < 8; ++r)
            lds[(q * 8 + r) * LDSP + ib * 16 + col] = (_Float16)hin[ib][r];
    asm volatile("s_wait_dscnt 0x0" ::: "memory");

    v16h a[4];
#pragma unroll
    for (int ks = 0; ks < 4; ++ks) a[ks] = load_a_f16(lds, LDSP, m, ks * 32, q);

    // pipeline B fragments one block deep
    v16h bf[4];
#pragma unroll
    for (int ks = 0; ks < 4; ++ks) bf[ks] = load_b(w, 128, ks * 32, 0, lane);

#pragma unroll
    for (int nb = 0; nb < 8; ++nb) {
        v16h cur[4];
#pragma unroll
        for (int ks = 0; ks < 4; ++ks) cur[ks] = bf[ks];
        if (nb < 7) {
#pragma unroll
            for (int ks = 0; ks < 4; ++ks)
                bf[ks] = load_b(w, 128, ks * 32, (nb + 1) * 16, lane);
        }
        v8f c = zero8();
#pragma unroll
        for (int ks = 0; ks < 4; ++ks) c = wmma16(a[ks], cur[ks], c);
        const float b = bias[nb * 16 + col];
#pragma unroll
        for (int r = 0; r < 8; ++r) hout[nb][r] = c[r] + b;
    }
    asm volatile("s_wait_dscnt 0x0" ::: "memory");
}

__device__ __forceinline__ void residual_step(v8f h[8], v8f t[8], _Float16* lds,
                                              const _Float16* w1, const float* b1,
                                              const _Float16* w2, const float* b2,
                                              int lane) {
    gemm128(h, t, lds, w1, b1, lane);
#pragma unroll
    for (int ib = 0; ib < 8; ++ib)
#pragma unroll
        for (int r = 0; r < 8; ++r) t[ib][r] = silu_f(t[ib][r]);
    gemm128(t, t, lds, w2, b2, lane);
#pragma unroll
    for (int ib = 0; ib < 8; ++ib)
#pragma unroll
        for (int r = 0; r < 8; ++r) h[ib][r] += silu_f(t[ib][r]);
}

// ---------------------------------------------------------------------------
// Kernel 5: h = x_ji + agg; residual(before); h = silu(h@lin+b)+x; 2x residual
// Entire chain is row-local: tile stays in registers through 7 GEMMs.
// ---------------------------------------------------------------------------
__global__ __launch_bounds__(128) void edge_post(
    const float* __restrict__ x, const float* __restrict__ x_ji,
    const float* __restrict__ agg, const _Float16* __restrict__ w3,
    const float* __restrict__ before_b, const float* __restrict__ lin_b,
    const float* __restrict__ after_b, float* __restrict__ out) {
    __shared__ _Float16 lds_all[4][16 * LDSP];
    const int lane = threadIdx.x & 31, wave = threadIdx.x >> 5;
    const int tile = blockIdx.x * 4 + wave;
    if (tile >= E_EDGES / 16) return;
    const int e0 = tile * 16;
    const int m = lane & 15, q = lane >> 4, col = m;
    _Float16* lds = &lds_all[wave][0];

    v8f h[8], t[8];
#pragma unroll
    for (int ib = 0; ib < 8; ++ib)
#pragma unroll
        for (int r = 0; r < 8; ++r) {
            size_t o = (size_t)(e0 + q * 8 + r) * 128 + ib * 16 + col;
            h[ib][r] = x_ji[o] + agg[o];
        }

    // before residual (1x)
    residual_step(h, t, lds, w3 + 0 * 16384, before_b + 0,
                  w3 + 1 * 16384, before_b + 128, lane);

    // h = silu(h @ lin_w + lin_b) + x
    gemm128(h, t, lds, w3 + 2 * 16384, lin_b, lane);
#pragma unroll
    for (int ib = 0; ib < 8; ++ib)
#pragma unroll
        for (int r = 0; r < 8; ++r) {
            size_t o = (size_t)(e0 + q * 8 + r) * 128 + ib * 16 + col;
            h[ib][r] = silu_f(t[ib][r]) + x[o];
        }

    // after residuals (2x)
    residual_step(h, t, lds, w3 + 3 * 16384, after_b + 0,
                  w3 + 4 * 16384, after_b + 128, lane);
    residual_step(h, t, lds, w3 + 5 * 16384, after_b + 256,
                  w3 + 6 * 16384, after_b + 384, lane);

#pragma unroll
    for (int ib = 0; ib < 8; ++ib)
#pragma unroll
        for (int r = 0; r < 8; ++r) {
            size_t o = (size_t)(e0 + q * 8 + r) * 128 + ib * 16 + col;
            out[o] = h[ib][r];
        }
}

// ---------------------------------------------------------------------------
extern "C" void kernel_launch(void* const* d_in, const int* in_sizes, int n_in,
                              void* d_out, int out_size, void* d_ws, size_t ws_size,
                              hipStream_t stream) {
    const float* x         = (const float*)d_in[0];
    const float* rbf       = (const float*)d_in[1];
    const float* sbf       = (const float*)d_in[2];
    const int*   idx_kj    = (const int*)d_in[3];
    const int*   idx_ji    = (const int*)d_in[4];
    const float* lin_rbf_w = (const float*)d_in[5];
    const float* lin_sbf_w = (const float*)d_in[6];
    const float* lin_kj_w  = (const float*)d_in[7];
    const float* lin_kj_b  = (const float*)d_in[8];
    const float* lin_ji_w  = (const float*)d_in[9];
    const float* lin_ji_b  = (const float*)d_in[10];
    const float* Wbig      = (const float*)d_in[11];
    const float* before_w  = (const float*)d_in[12];
    const float* before_b  = (const float*)d_in[13];
    const float* lin_w     = (const float*)d_in[14];
    const float* lin_b     = (const float*)d_in[15];
    const float* after_w   = (const float*)d_in[16];
    const float* after_b   = (const float*)d_in[17];

    char* ws = (char*)d_ws;
    float*    x_ji  = (float*)ws;    ws += (size_t)E_EDGES * 128 * 4;
    _Float16* x_kj  = (_Float16*)ws; ws += (size_t)E_EDGES * 128 * 2;
    float*    agg   = (float*)ws;    ws += (size_t)E_EDGES * 128 * 4;
    float*    rbf_p = (float*)ws;    ws += (size_t)E_EDGES * 128 * 4;
    float*    sbf_p = (float*)ws;    ws += (size_t)T_TRIP * 8 * 4;
    _Float16* wji16 = (_Float16*)ws; ws += 16384 * 2;
    _Float16* wkj16 = (_Float16*)ws; ws += 16384 * 2;
    _Float16* Wr16  = (_Float16*)ws; ws += 131072 * 2;
    _Float16* w3    = (_Float16*)ws; ws += 7 * 16384 * 2;

    prep_weights<<<(131072 + 255) / 256, 256, 0, stream>>>(
        lin_ji_w, lin_kj_w, Wbig, before_w, lin_w, after_w,
        wji16, wkj16, Wr16, w3);

    proj_small<<<(E_EDGES * 128 + 255) / 256, 256, 0, stream>>>(
        rbf, lin_rbf_w, sbf, lin_sbf_w, rbf_p, sbf_p, agg);

    edge_pre<<<(E_EDGES / 16 + 3) / 4, 128, 0, stream>>>(
        x, rbf_p, wji16, wkj16, lin_ji_b, lin_kj_b, x_ji, x_kj);

    triplet_gemm<<<(T_TRIP / 16 + 3) / 4, 128, 0, stream>>>(
        x_kj, Wr16, sbf_p, idx_kj, idx_ji, agg);

    edge_post<<<(E_EDGES / 16 + 3) / 4, 128, 0, stream>>>(
        x, x_ji, agg, w3, before_b, lin_b, after_b, (float*)d_out);
}
